// FeRALinear_10763188044294
// MI455X (gfx1250) — matrix operand order repaired and checked by reading, not compiled, not run on hardware
//
#include <hip/hip_runtime.h>
#include <hip/hip_bf16.h>

// ---------------------------------------------------------------------------
// FeRA linear: out = x @ W^T + sum_n routing[b,n] * (x @ Ld_n^T) @ Lu_n^T * SCALE
// bf16 WMMA GEMMs on gfx1250, tiles staged to LDS by the Tensor Data Mover
// (tensor_load_to_lds, TENSORcnt), double-buffered.
//   K1: base   [8192x4096] = xb * Wb^T                              -> d_out f32
//   K2: down   [8192x128]  = xb * Ad^T, scaled by routing           -> ws bf16
//   K3: up     d_out      += down * U^T   (K = N_EXP*RANK = 128)
// ---------------------------------------------------------------------------

typedef __attribute__((ext_vector_type(16))) __bf16 v16bf;
typedef __attribute__((ext_vector_type(8)))  float  v8f;
typedef __attribute__((ext_vector_type(4)))  unsigned int u32x4;
typedef __attribute__((ext_vector_type(8)))  unsigned int u32x8;

struct Frag32 { uint4 lo, hi; };   // 32 bytes == v16bf

#define B_Q    4
#define S_Q    2048
#define D_IN   4096
#define D_OUT  4096
#define N_EXP  8
#define RANK_Q 16
#define SCALE_Q 2.0f   // ALPHA / RANK = 32/16

#define TILE_BYTES 10240   // 128 rows * 80B padded pitch (64B data + 16B pad)

__device__ __forceinline__ unsigned short f32_to_bf16_rne(float f) {
  unsigned int u = __float_as_uint(f);
  unsigned int r = u + 0x7FFFu + ((u >> 16) & 1u);   // round-to-nearest-even
  return (unsigned short)(r >> 16);
}

// ---------------------------------------------------------------------------
// elementwise f32 -> bf16 (4 per thread, vectorized)
// ---------------------------------------------------------------------------
__global__ __launch_bounds__(256)
void cvt4_f32_bf16(const float* __restrict__ s, unsigned short* __restrict__ d, int n) {
  int i = (blockIdx.x * 256 + threadIdx.x) * 4;
  if (i + 4 <= n) {
    float4 f = *(const float4*)(s + i);
    uint2 o;
    o.x = (unsigned)f32_to_bf16_rne(f.x) | ((unsigned)f32_to_bf16_rne(f.y) << 16);
    o.y = (unsigned)f32_to_bf16_rne(f.z) | ((unsigned)f32_to_bf16_rne(f.w) << 16);
    *(uint2*)(d + i) = o;
  }
}

// gather lora_up [N, D_out, R] -> U [D_out, N*R] bf16
__global__ __launch_bounds__(256)
void build_up_kernel(const float* __restrict__ lup, unsigned short* __restrict__ U) {
  int idx = blockIdx.x * 256 + threadIdx.x;
  if (idx < D_OUT * N_EXP * RANK_Q) {
    int o  = idx >> 7;         // / 128
    int j  = idx & 127;
    int n  = j >> 4;
    int rr = j & 15;
    U[idx] = f32_to_bf16_rne(lup[(n * D_OUT + o) * RANK_Q + rr]);
  }
}

// ---------------------------------------------------------------------------
// TDM: DMA one 128x32 bf16 tile (row-major, leading dim `ld` elements) from
// global into LDS at `lds_addr`, padding each 64B row to an 80B pitch.
// 2D tensor -> only D# groups 0 and 1 are needed.
// ---------------------------------------------------------------------------
__device__ __forceinline__ void tdm_load_tile(const unsigned short* gptr,
                                              unsigned lds_addr,
                                              int ld, int rows) {
  unsigned long long ga = (unsigned long long)(size_t)(const void*)gptr;
  u32x4 g0;
  g0[0] = 1u;                                              // count=1 (valid D#)
  g0[1] = lds_addr;                                        // lds_addr [63:32]
  g0[2] = (unsigned)ga;                                    // global_addr lo
  g0[3] = (unsigned)((ga >> 32) & 0x01FFFFFFu)             // global_addr hi
        | 0x80000000u;                                     // type=2 ("image")
  u32x8 g1;
  // data_size=1 (2B) | pad_enable | pad_interval=3 (16 DW) | pad_amount=3 (4 DW)
  g1[0] = 0x06D10000u;
  g1[1] = ((unsigned)ld & 0xFFFFu) << 16;                  // tensor_dim0 lo
  g1[2] = ((unsigned)ld >> 16) | (((unsigned)rows & 0xFFFFu) << 16); // td0 hi | td1 lo
  g1[3] = ((unsigned)rows >> 16) | (32u << 16);            // td1 hi | tile_dim0=32
  g1[4] = 128u;                                            // tile_dim1=128, tile_dim2=0
  g1[5] = (unsigned)ld;                                    // tensor_dim0_stride lo
  g1[6] = 0u;                                              // stride0 hi | stride1 lo
  g1[7] = 0u;
  asm volatile("tensor_load_to_lds %0, %1" :: "s"(g0), "s"(g1) : "memory");
}

// ---------------------------------------------------------------------------
// Tiled WMMA GEMM: C[m,n] = sum_k A[m,k] * B[n,k]   (both row-major over K)
// Block tile 128x128, BK=32, 256 threads = 8 wave32 waves, wave tile 64x32.
// A/B tiles DMA'd into double-buffered LDS by the TDM (waves 0 and 1 issue).
// MODE 0: store f32 to Cf             (base GEMM)
// MODE 1: scale by SCALE*routing, store bf16 to Cbf, stride 128 (down GEMM)
// MODE 2: Cf[idx] += acc              (up GEMM, accumulate onto base)
// ---------------------------------------------------------------------------
template<int MODE>
__global__ __launch_bounds__(256)
void fera_gemm(const unsigned short* __restrict__ A,
               const unsigned short* __restrict__ Bm,
               int K, int lda, int ldb, int rowsB,
               float* __restrict__ Cf, int ldc,
               unsigned short* __restrict__ Cbf,
               const float* __restrict__ routing)
{
  // two buffers per matrix; padded rows: 32 data + 8 pad elems (80B pitch)
  __shared__ alignas(16) unsigned short lA[2][128 * 40];
  __shared__ alignas(16) unsigned short lB[2][128 * 40];

  const int m0   = blockIdx.y * 128;
  const int n0   = blockIdx.x * 128;
  const int t    = threadIdx.x;
  const int lane = t & 31;
  const int wave = t >> 5;
  const int wm   = (wave >> 2) * 64;  // wave M offset in tile: 0 / 64
  const int wn   = (wave & 3) * 32;   // wave N offset in tile: 0..96
  const int r15  = lane & 15;
  const int kh   = lane >> 4;         // half-wave selector

  const unsigned ldsA0 = (unsigned)(unsigned long long)(size_t)(void*)&lA[0][0];
  const unsigned ldsB0 = (unsigned)(unsigned long long)(size_t)(void*)&lB[0][0];
  const unsigned short* Arow = A  + (size_t)m0 * lda;   // tile row origin
  const unsigned short* Brow = Bm + (size_t)n0 * ldb;

  v8f acc[4][2];
#pragma unroll
  for (int mi = 0; mi < 4; ++mi)
#pragma unroll
    for (int ni = 0; ni < 2; ++ni)
#pragma unroll
      for (int v = 0; v < 8; ++v) acc[mi][ni][v] = 0.0f;

  // prologue: DMA first k-slab into buffer 0
  if (wave == 0)      tdm_load_tile(Arow, ldsA0, lda, 8192);
  else if (wave == 1) tdm_load_tile(Brow, ldsB0, ldb, rowsB);

  int buf = 0;
  for (int k0 = 0; k0 < K; k0 += 32) {
    // issue DMA for next slab into the other buffer, then wait for current
    if (k0 + 32 < K) {
      if (wave == 0)      tdm_load_tile(Arow + (k0 + 32), ldsA0 + (buf ^ 1) * TILE_BYTES, lda, 8192);
      else if (wave == 1) tdm_load_tile(Brow + (k0 + 32), ldsB0 + (buf ^ 1) * TILE_BYTES, ldb, rowsB);
      if (wave < 2) __builtin_amdgcn_s_wait_tensorcnt(1);  // in-order: current slab done
    } else {
      if (wave < 2) __builtin_amdgcn_s_wait_tensorcnt(0);
    }
    __syncthreads();   // publish current buffer to all waves

    const unsigned short* sA = lA[buf];
    const unsigned short* sB = lB[buf];

    // A frag (16x32): lane = M row; vec[0..7]=K(kh*8..), vec[8..15]=K(16+kh*8..)
    v16bf aF[4];
#pragma unroll
    for (int mi = 0; mi < 4; ++mi) {
      const int row = wm + mi * 16 + r15;
      Frag32 f;
      f.lo = *(const uint4*)&sA[row * 40 + kh * 8];
      f.hi = *(const uint4*)&sA[row * 40 + 16 + kh * 8];
      aF[mi] = __builtin_bit_cast(v16bf, f);
    }
    // B frag (32x16): lane = N column; vec[0..7]=K(kh*8..), vec[8..15]=K(16+kh*8..)
    v16bf bF[2];
#pragma unroll
    for (int ni = 0; ni < 2; ++ni) {
      const int row = wn + ni * 16 + r15;
      const int kb  = kh * 8;
      Frag32 f;
      f.lo = *(const uint4*)&sB[row * 40 + kb];
      f.hi = *(const uint4*)&sB[row * 40 + kb + 16];
      bF[ni] = __builtin_bit_cast(v16bf, f);
    }

#pragma unroll
    for (int mi = 0; mi < 4; ++mi)
#pragma unroll
      for (int ni = 0; ni < 2; ++ni)
        acc[mi][ni] = __builtin_amdgcn_wmma_f32_16x16x32_bf16(
            false, aF[mi], false, bF[ni], (short)0, acc[mi][ni], false, false);

    __syncthreads();   // all reads of `buf` done before TDM overwrites it (k0+64)
    buf ^= 1;
  }

  // epilogue: C layout — VGPR v: M = v + 8*(lane>>4), N = lane&15
  const int madd = kh * 8;
#pragma unroll
  for (int mi = 0; mi < 4; ++mi) {
    const int mb = m0 + wm + mi * 16 + madd;
#pragma unroll
    for (int ni = 0; ni < 2; ++ni) {
      const int nn = n0 + wn + ni * 16 + r15;
      if constexpr (MODE == 0) {
#pragma unroll
        for (int v = 0; v < 8; ++v)
          Cf[(mb + v) * ldc + nn] = acc[mi][ni][v];
      } else if constexpr (MODE == 1) {
        const int b = m0 / S_Q;                          // 128-row tile never crosses b
        const float sc = SCALE_Q * routing[b * N_EXP + (nn >> 4)];
#pragma unroll
        for (int v = 0; v < 8; ++v)
          Cbf[(mb + v) * 128 + nn] = f32_to_bf16_rne(acc[mi][ni][v] * sc);
      } else {
#pragma unroll
        for (int v = 0; v < 8; ++v)
          Cf[(mb + v) * ldc + nn] += acc[mi][ni][v];
      }
    }
  }
}

// ---------------------------------------------------------------------------
extern "C" void kernel_launch(void* const* d_in, const int* in_sizes, int n_in,
                              void* d_out, int out_size, void* d_ws, size_t ws_size,
                              hipStream_t stream) {
  (void)in_sizes; (void)n_in; (void)out_size; (void)ws_size;

  const float* x       = (const float*)d_in[0];   // [4,2048,4096]
  const float* routing = (const float*)d_in[1];   // [4,8]
  const float* Wf      = (const float*)d_in[2];   // [4096,4096]
  const float* ldown   = (const float*)d_in[3];   // [8,16,4096] == [128,4096]
  const float* lup     = (const float*)d_in[4];   // [8,4096,16]
  float* out = (float*)d_out;                     // [4,2048,4096]

  // workspace layout (bf16 elements)
  unsigned short* xb  = (unsigned short*)d_ws;    // 33,554,432  (x bf16)
  unsigned short* Wb  = xb  + 33554432;           // 16,777,216  (W bf16)
  unsigned short* Ad  = Wb  + 16777216;           //    524,288  (lora_down stack [128,4096])
  unsigned short* Ub  = Ad  + 524288;             //    524,288  (lora_up gathered [4096,128])
  unsigned short* dwn = Ub  + 524288;             //  1,048,576  (scaled down [8192,128])

  const int M = B_Q * S_Q;                        // 8192

  // 1) precision conversion / repacking
  cvt4_f32_bf16<<<dim3((M * D_IN) / 1024), dim3(256), 0, stream>>>(x, xb, M * D_IN);
  cvt4_f32_bf16<<<dim3((D_OUT * D_IN) / 1024), dim3(256), 0, stream>>>(Wf, Wb, D_OUT * D_IN);
  cvt4_f32_bf16<<<dim3((N_EXP * RANK_Q * D_IN) / 1024), dim3(256), 0, stream>>>(ldown, Ad, N_EXP * RANK_Q * D_IN);
  build_up_kernel<<<dim3((D_OUT * N_EXP * RANK_Q) / 256), dim3(256), 0, stream>>>(lup, Ub);

  // 2) base GEMM: out = xb * Wb^T          (8192 x 4096 x 4096)
  fera_gemm<0><<<dim3(D_OUT / 128, M / 128), dim3(256), 0, stream>>>(
      xb, Wb, D_IN, D_IN, D_IN, D_OUT, out, D_OUT, nullptr, nullptr);

  // 3) down GEMM: dwn = scale * (xb * Ad^T)  (8192 x 128 x 4096)
  fera_gemm<1><<<dim3(1, M / 128), dim3(256), 0, stream>>>(
      xb, Ad, D_IN, D_IN, D_IN, N_EXP * RANK_Q, nullptr, 0, dwn, routing);

  // 4) up GEMM: out += dwn * Ub^T          (8192 x 4096 x 128)
  fera_gemm<2><<<dim3(D_OUT / 128, M / 128), dim3(256), 0, stream>>>(
      dwn, Ub, N_EXP * RANK_Q, N_EXP * RANK_Q, N_EXP * RANK_Q, D_OUT, out, D_OUT, nullptr, nullptr);
}
